// ResBlock_2_77137612636480
// MI455X (gfx1250) — 8-bit matrix operand order repaired and checked by reading, compiled, not run on hardware
//
#include <hip/hip_runtime.h>
#include <stdint.h>
#include <stddef.h>

typedef int v8i  __attribute__((ext_vector_type(8)));
typedef int v4i_t __attribute__((ext_vector_type(4)));

union U8i {
    v8i v;
    v4i_t h[2];
    long long q[4];
    int d[8];
};

#if defined(__has_builtin)
# if __has_builtin(__builtin_amdgcn_global_load_async_to_lds_b128) && \
     __has_builtin(__builtin_amdgcn_s_wait_asynccnt)
#  define USE_ASYNC_LDS 1
# endif
#endif
#ifndef USE_ASYNC_LDS
# define USE_ASYNC_LDS 0
#endif

typedef __attribute__((address_space(1))) v4i_t gv4i_as1;
typedef __attribute__((address_space(3))) v4i_t lv4i_as3;

#define OFF_XB      ((size_t)0)
#define OFF_Y1B     ((size_t)25690112)
#define OFF_B1      ((size_t)38535168)
#define OFF_B2      ((size_t)39714816)
#define OFF_BS      ((size_t)42074112)
#define OFF_Y1RAW   ((size_t)42205184)
#define OFF_Y2RAW   ((size_t)93585408)
#define OFF_SRAW    ((size_t)144965632)
#define OFF_STATS   ((size_t)196345856)

#define COUT 512
#define MPIX 25088
#define BN_COUNT 25088.0f

__global__ void zero_f32(float* __restrict__ p, int n) {
    int i = blockIdx.x * blockDim.x + threadIdx.x;
    if (i < n) p[i] = 0.0f;
}

__global__ void binarize_x_kernel(const float* __restrict__ x, char* __restrict__ xb,
                                  int C, int H, int W, size_t total) {
    size_t idx = (size_t)blockIdx.x * blockDim.x + threadIdx.x;
    if (idx >= total) return;
    int c = (int)(idx % C);  size_t t = idx / C;
    int w = (int)(t % W);    t /= W;
    int h = (int)(t % H);    int n = (int)(t / H);
    float v = x[(((size_t)n * C + c) * H + h) * W + w];
    xb[idx] = (v > 0.0f) ? (char)1 : ((v < 0.0f) ? (char)-1 : (char)0);
}

__global__ void binarize_w_kernel(const float* __restrict__ w, char* __restrict__ Bw,
                                  int Cin, int KHW, size_t total) {
    size_t idx = (size_t)blockIdx.x * blockDim.x + threadIdx.x;
    if (idx >= total) return;
    int k = (int)(idx % KHW); size_t t = idx / KHW;
    int c = (int)(t % Cin);   int cout = (int)(t / Cin);
    float v = w[idx];
    char s = (v > 0.0f) ? (char)1 : ((v < 0.0f) ? (char)-1 : (char)0);
    int K = k * Cin + c;
    Bw[(((size_t)(K >> 6) * COUT + cout) << 6) + (K & 63)] = s;
}

__device__ __forceinline__ void stage16(const char* gsrc, char* ldst) {
#if USE_ASYNC_LDS
    __builtin_amdgcn_global_load_async_to_lds_b128(
        (gv4i_as1*)(uintptr_t)gsrc,
        (lv4i_as3*)(uint32_t)(uintptr_t)ldst, 0, 0);
#else
    *(v4i_t*)ldst = *(const v4i_t*)gsrc;
#endif
}

__global__ __launch_bounds__(256) void bconv_wmma_kernel(
    const char* __restrict__ xin, const char* __restrict__ Bw,
    float* __restrict__ yraw, float* __restrict__ gsum, float* __restrict__ gsq,
    int IH, int IW, int Cin, int OH, int OW, int KH, int KW, int stride, int pad)
{
    __shared__ __align__(16) char Btile[2][64 * 64];
    __shared__ float chsum[64];
    __shared__ float chsq[64];

    const int tid  = threadIdx.x;
    const int lane = tid & 31;
    const int wv   = tid >> 5;
    const int l16  = lane & 15;
    const int hi   = lane >> 4;

    if (tid < 64) { chsum[tid] = 0.0f; chsq[tid] = 0.0f; }

    const int nbase = blockIdx.y * 64;
    const int m0    = blockIdx.x * 128 + wv * 16;
    const int m     = m0 + l16;
    const int HW    = OH * OW;
    const int n_img = m / HW;
    const int rem   = m % HW;
    const int oh    = rem / OW;
    const int ow    = rem % OW;

    const int k0  = hi ? 16 : 0;
    const int klo = hi ? 16 : 0;

    U8i acc[4];
#pragma unroll
    for (int j = 0; j < 4; ++j)
#pragma unroll
        for (int r = 0; r < 8; ++r) acc[j].d[r] = 0;

    const int CB = Cin >> 6;
    const int KB = KH * KW * CB;

    stage16(Bw + ((size_t)nbase) * 64 + tid * 16, &Btile[0][tid * 16]);
#if USE_ASYNC_LDS
    __builtin_amdgcn_s_wait_asynccnt(0);
#endif
    __syncthreads();

    for (int kb = 0; kb < KB; ++kb) {
        const int buf  = kb & 1;
        const int kidx = kb / CB;
        const int cb   = kb - kidx * CB;
        const int ky   = kidx / KW;
        const int kx   = kidx - ky * KW;

        const int ihh = oh * stride + ky - pad;
        const int iww = ow * stride + kx - pad;
        const bool vld = (ihh >= 0) && (ihh < IH) && (iww >= 0) && (iww < IW);
        const int ihc = min(max(ihh, 0), IH - 1);
        const int iwc = min(max(iww, 0), IW - 1);
        const char* ap =
            xin + ((size_t)((n_img * IH + ihc) * IW + iwc)) * Cin + (cb << 6) + k0;

        long long q0 = *(const long long*)(ap);
        long long q1 = *(const long long*)(ap + 8);
        long long q2 = *(const long long*)(ap + 32);
        long long q3 = *(const long long*)(ap + 40);
        U8i a;
        a.q[0] = vld ? q0 : 0ll;
        a.q[1] = vld ? q1 : 0ll;
        a.q[2] = vld ? q2 : 0ll;
        a.q[3] = vld ? q3 : 0ll;

        const bool has_next = (kb + 1) < KB;
        v4i_t pre;
#if USE_ASYNC_LDS
        if (has_next)
            stage16(Bw + ((size_t)((kb + 1) * COUT + nbase)) * 64 + tid * 16,
                    &Btile[buf ^ 1][tid * 16]);
#else
        if (has_next)
            pre = *(const v4i_t*)(Bw + ((size_t)((kb + 1) * COUT + nbase)) * 64 + tid * 16);
#endif

#pragma unroll
        for (int j = 0; j < 4; ++j) {
            U8i b;
            const char* bp = &Btile[buf][((j * 16 + l16) << 6) + klo];
            b.h[0] = *(const v4i_t*)(bp);
            b.h[1] = *(const v4i_t*)(bp + 32);
            acc[j].v = __builtin_amdgcn_wmma_i32_16x16x64_iu8(
                true, a.v, true, b.v, acc[j].v, false, false);
        }

        if (has_next) {
#if USE_ASYNC_LDS
            __builtin_amdgcn_s_wait_asynccnt(0);
#else
            *(v4i_t*)(&Btile[buf ^ 1][tid * 16]) = pre;
#endif
        }
        __syncthreads();
    }

    const int mrow = m0 + (hi << 3);
#pragma unroll
    for (int j = 0; j < 4; ++j) {
        const int coll = j * 16 + l16;
        const int col  = nbase + coll;
        float s = 0.0f, qq = 0.0f;
#pragma unroll
        for (int r = 0; r < 8; ++r) {
            float f = (float)acc[j].d[r];
            yraw[(size_t)(mrow + r) * COUT + col] = f;
            s += f; qq += f * f;
        }
        atomicAdd(&chsum[coll], s);
        atomicAdd(&chsq[coll], qq);
    }
    __syncthreads();
    if (tid < 64) {
        atomicAdd(&gsum[nbase + tid], chsum[tid]);
        atomicAdd(&gsq[nbase + tid],  chsq[tid]);
    }
}

__global__ void bn_finalize_kernel(const float* __restrict__ sum, const float* __restrict__ sq,
                                   const float* __restrict__ gamma, const float* __restrict__ beta,
                                   float* __restrict__ scale, float* __restrict__ shift,
                                   float inv_count) {
    int c = blockIdx.x * blockDim.x + threadIdx.x;
    if (c >= COUT) return;
    float mean = sum[c] * inv_count;
    float var  = sq[c] * inv_count - mean * mean;
    float sc   = gamma[c] * rsqrtf(var + 1e-5f);
    scale[c] = sc;
    shift[c] = beta[c] - mean * sc;
}

__global__ void binarize_bn_kernel(const float* __restrict__ y, const float* __restrict__ scale,
                                   const float* __restrict__ shift, char* __restrict__ out,
                                   size_t total) {
    size_t idx = (size_t)blockIdx.x * blockDim.x + threadIdx.x;
    if (idx >= total) return;
    int c = (int)(idx & (COUT - 1));
    float v = y[idx] * scale[c] + shift[c];
    out[idx] = (v > 0.0f) ? (char)1 : ((v < 0.0f) ? (char)-1 : (char)0);
}

__global__ void final_add_kernel(const float* __restrict__ sraw, const float* __restrict__ y2,
                                 const float* __restrict__ sc_s, const float* __restrict__ sh_s,
                                 const float* __restrict__ sc_y, const float* __restrict__ sh_y,
                                 float* __restrict__ out, int OH, int OW, size_t total) {
    size_t idx = (size_t)blockIdx.x * blockDim.x + threadIdx.x;
    if (idx >= total) return;
    int w = (int)(idx % OW);             size_t t = idx / OW;
    int h = (int)(t % OH);               t /= OH;
    int c = (int)(t % COUT);             int n = (int)(t / COUT);
    size_t p = ((size_t)((n * OH + h) * OW + w)) * COUT + c;
    out[idx] = (sraw[p] * sc_s[c] + sh_s[c]) + (y2[p] * sc_y[c] + sh_y[c]);
}

extern "C" void kernel_launch(void* const* d_in, const int* in_sizes, int n_in,
                              void* d_out, int out_size, void* d_ws, size_t ws_size,
                              hipStream_t stream) {
    const float* x  = (const float*)d_in[0];
    const float* w1 = (const float*)d_in[1];
    const float* g1 = (const float*)d_in[2];
    const float* b1 = (const float*)d_in[3];
    const float* w2 = (const float*)d_in[4];
    const float* g2 = (const float*)d_in[5];
    const float* b2 = (const float*)d_in[6];
    const float* ws = (const float*)d_in[7];
    const float* gs = (const float*)d_in[8];
    const float* bs = (const float*)d_in[9];

    char* base = (char*)d_ws;
    char*  xb    = base + OFF_XB;
    char*  y1b   = base + OFF_Y1B;
    char*  B1    = base + OFF_B1;
    char*  B2    = base + OFF_B2;
    char*  Bs    = base + OFF_BS;
    float* y1raw = (float*)(base + OFF_Y1RAW);
    float* y2raw = (float*)(base + OFF_Y2RAW);
    float* sraw  = (float*)(base + OFF_SRAW);
    float* st    = (float*)(base + OFF_STATS);
    float* sum1 = st + 0 * COUT;  float* sq1 = st + 1 * COUT;
    float* sum2 = st + 2 * COUT;  float* sq2 = st + 3 * COUT;
    float* sums = st + 4 * COUT;  float* sqs = st + 5 * COUT;
    float* sc1  = st + 6 * COUT;  float* sh1 = st + 7 * COUT;
    float* sc2  = st + 8 * COUT;  float* sh2 = st + 9 * COUT;
    float* scs  = st + 10 * COUT; float* shs = st + 11 * COUT;

    const float inv_cnt = 1.0f / BN_COUNT;

    zero_f32<<<dim3(12), dim3(256), 0, stream>>>(st, 6 * COUT);

    {
        size_t total = (size_t)32 * 56 * 56 * 256;
        binarize_x_kernel<<<dim3((unsigned)((total + 255) / 256)), dim3(256), 0, stream>>>(
            x, xb, 256, 56, 56, total);
    }
    binarize_w_kernel<<<dim3(4608), dim3(256), 0, stream>>>(w1, B1, 256, 9, (size_t)512 * 256 * 9);
    binarize_w_kernel<<<dim3(9216), dim3(256), 0, stream>>>(w2, B2, 512, 9, (size_t)512 * 512 * 9);
    binarize_w_kernel<<<dim3(512),  dim3(256), 0, stream>>>(ws, Bs, 256, 1, (size_t)512 * 256);

    const dim3 cgrid(MPIX / 128, COUT / 64);
    bconv_wmma_kernel<<<cgrid, dim3(256), 0, stream>>>(
        xb, B1, y1raw, sum1, sq1, 56, 56, 256, 28, 28, 3, 3, 2, 1);
    bconv_wmma_kernel<<<cgrid, dim3(256), 0, stream>>>(
        xb, Bs, sraw, sums, sqs, 56, 56, 256, 28, 28, 1, 1, 2, 0);

    bn_finalize_kernel<<<dim3(2), dim3(256), 0, stream>>>(sum1, sq1, g1, b1, sc1, sh1, inv_cnt);
    {
        size_t total = (size_t)MPIX * COUT;
        binarize_bn_kernel<<<dim3((unsigned)((total + 255) / 256)), dim3(256), 0, stream>>>(
            y1raw, sc1, sh1, y1b, total);
    }

    bconv_wmma_kernel<<<cgrid, dim3(256), 0, stream>>>(
        y1b, B2, y2raw, sum2, sq2, 28, 28, 512, 28, 28, 3, 3, 1, 1);

    bn_finalize_kernel<<<dim3(2), dim3(256), 0, stream>>>(sum2, sq2, g2, b2, sc2, sh2, inv_cnt);
    bn_finalize_kernel<<<dim3(2), dim3(256), 0, stream>>>(sums, sqs, gs, bs, scs, shs, inv_cnt);

    {
        size_t total = (size_t)MPIX * COUT;
        final_add_kernel<<<dim3((unsigned)((total + 255) / 256)), dim3(256), 0, stream>>>(
            sraw, y2raw, scs, shs, sc2, sh2, (float*)d_out, 28, 28, total);
    }
}
